// TopKAuto_31267361915196
// MI455X (gfx1250) — matrix operand order repaired and checked by reading, not compiled, not run on hardware
//
#include <hip/hip_runtime.h>

// ---------------------------------------------------------------------------
// TopK sparse autoencoder forward, MI455X (gfx1250, wave32, WMMA bf16)
//   B=4096, S=3, D=1024 (SD=3072), H=16384, K=64
// ---------------------------------------------------------------------------

typedef __attribute__((ext_vector_type(16))) __bf16 v16bf;
typedef __attribute__((ext_vector_type(8)))  __bf16 v8bf;
typedef __attribute__((ext_vector_type(8)))  float  v8f;
typedef __attribute__((ext_vector_type(4)))  float  v4f;

#define BATCH 4096
#define SRC   3
#define DIN   1024
#define SD    3072      // S*D, GEMM K dimension
#define HID   16384     // GEMM N dimension
#define TOPK  64

// GEMM tiling
#define BM 128
#define BN 128
#define BK 32
#define LDS_STR 40      // padded row stride (elements) -> 80B, conflict-free b128

// ---------------------------------------------------------------------------
// fp32 -> bf16 conversion (vectorized x4)
// ---------------------------------------------------------------------------
__global__ __launch_bounds__(256) void cvt_f32_bf16(const float* __restrict__ in,
                                                    __bf16* __restrict__ out,
                                                    size_t n4) {
  size_t i = (size_t)blockIdx.x * blockDim.x + threadIdx.x;
  size_t stride = (size_t)gridDim.x * blockDim.x;
  for (; i < n4; i += stride) {
    v4f v = *(const v4f*)(in + 4 * i);
    v8bf o;  // unused; results are stored per element below
    __bf16 b0 = (__bf16)v.x, b1 = (__bf16)v.y, b2 = (__bf16)v.z, b3 = (__bf16)v.w;
    __bf16* dst = out + 4 * i;
    dst[0] = b0; dst[1] = b1; dst[2] = b2; dst[3] = b3;
    (void)o;
  }
}

// ---------------------------------------------------------------------------
// Encode GEMM: latents[b][h] = sum_k xb[b][k] * wb[k][h] + b_enc[h]
//   xb: bf16 [BATCH][SD] row-major, wb: bf16 [SD][HID] row-major
//   block tile 128x128, 8 waves of 32(M)x64(N), v_wmma_f32_16x16x32_bf16
// ---------------------------------------------------------------------------
__global__ __launch_bounds__(256) void encode_gemm(const __bf16* __restrict__ xb,
                                                   const __bf16* __restrict__ wb,
                                                   const float* __restrict__ b_enc,
                                                   float* __restrict__ latents) {
  __shared__ __align__(16) __bf16 lX[BM * LDS_STR];  // [M][K] padded
  __shared__ __align__(16) __bf16 lW[BN * LDS_STR];  // [N][K] padded (transposed)

  const int t    = threadIdx.x;
  const int lane = t & 31;
  const int wave = t >> 5;
  const int n0   = blockIdx.x * BN;
  const int m0   = blockIdx.y * BM;
  const int wm   = (wave & 3) * 32;   // wave M offset in block tile
  const int wn   = (wave >> 2) * 64;  // wave N offset in block tile

  v8f acc[2][4] = {};

  const int lh  = lane & 15;          // 0..15
  const int lq  = lane >> 4;          // 0 or 1
  const int kbA = lq * 8;             // A frag: lane-half K split at 8
  const int kbB = lq * 8;             // B frag: lane-half K split at 8

  for (int k0 = 0; k0 < SD; k0 += BK) {
    // --- stage X tile [128][32] ---
#pragma unroll
    for (int i = 0; i < 2; ++i) {
      int c = t + i * 256;                    // 512 chunks of 8 bf16
      int r = c >> 2, col = (c & 3) * 8;
      v8bf v = *(const v8bf*)(xb + (size_t)(m0 + r) * SD + k0 + col);
      *(v8bf*)(&lX[r * LDS_STR + col]) = v;
    }
    // --- stage W tile [32][128] transposed into LDS [n][k] ---
#pragma unroll
    for (int i = 0; i < 2; ++i) {
      int c  = t + i * 256;                   // 512 chunks of 8 along n
      int kk = c >> 4;                        // 0..31
      int nn = (c & 15) * 8;                  // 0..120
      v8bf v = *(const v8bf*)(wb + (size_t)(k0 + kk) * HID + n0 + nn);
#pragma unroll
      for (int j = 0; j < 8; ++j) lW[(nn + j) * LDS_STR + kk] = v[j];
    }
    // prefetch next tiles into cache hierarchy while we compute
    if (k0 + BK < SD) {
      __builtin_prefetch(wb + (size_t)(k0 + BK) * HID + n0 + (t & 127), 0, 1);
      __builtin_prefetch(xb + (size_t)(m0 + (t & 127)) * SD + k0 + BK, 0, 1);
    }
    __syncthreads();

    // --- A fragments (16x32 bf16, ISA 7.12.2 layout) ---
    v16bf afrag[2];
#pragma unroll
    for (int mt = 0; mt < 2; ++mt) {
      const __bf16* rowp = &lX[(wm + mt * 16 + lh) * LDS_STR];
      v8bf alo = *(const v8bf*)(rowp + kbA);
      v8bf ahi = *(const v8bf*)(rowp + 16 + kbA);
      afrag[mt] = __builtin_shufflevector(alo, ahi, 0, 1, 2, 3, 4, 5, 6, 7,
                                          8, 9, 10, 11, 12, 13, 14, 15);
    }
    // --- B fragments (32x16 bf16; per-lane column, K split at 8 like A) + WMMA ---
#pragma unroll
    for (int nt = 0; nt < 4; ++nt) {
      const __bf16* colp = &lW[(wn + nt * 16 + lh) * LDS_STR];
      v8bf blo = *(const v8bf*)(colp + kbB);
      v8bf bhi = *(const v8bf*)(colp + 16 + kbB);
      v16bf bfrag = __builtin_shufflevector(blo, bhi, 0, 1, 2, 3, 4, 5, 6, 7,
                                            8, 9, 10, 11, 12, 13, 14, 15);
#pragma unroll
      for (int mt = 0; mt < 2; ++mt) {
        acc[mt][nt] = __builtin_amdgcn_wmma_f32_16x16x32_bf16(
            false, afrag[mt], false, bfrag, (short)0, acc[mt][nt], false, false);
      }
    }
    __syncthreads();
  }

  // --- epilogue: D layout (ISA 7.12.2): VGPR r, lane L -> M=r+8*(L/16), N=L%16
#pragma unroll
  for (int nt = 0; nt < 4; ++nt) {
    int Ncol = n0 + wn + nt * 16 + lh;
    float be = b_enc[Ncol];
#pragma unroll
    for (int mt = 0; mt < 2; ++mt) {
      int Mbase = m0 + wm + mt * 16 + lq * 8;
#pragma unroll
      for (int r = 0; r < 8; ++r) {
        latents[(size_t)(Mbase + r) * HID + Ncol] = acc[mt][nt][r] + be;
      }
    }
  }
}

// ---------------------------------------------------------------------------
// Top-64 per row with ReLU. One block (256 threads) per batch row.
// Deterministic: stable tie-break (equal value -> smaller index wins),
// fixed-shape tree reduction. Destroys latents (regenerated every call).
// ---------------------------------------------------------------------------
__global__ __launch_bounds__(256) void topk_kernel(float* __restrict__ latents,
                                                   float* __restrict__ topv,
                                                   int* __restrict__ topi) {
  __shared__ float cv[256];
  __shared__ int   ci[256];
  __shared__ int   sWin;
  const int b = blockIdx.x;
  const int t = threadIdx.x;
  float* row = latents + (size_t)b * HID;
  const int CH = HID / 256;  // 64 elements per thread

  float mv = -__builtin_inff();
  int   mi = t * CH;
  for (int j = 0; j < CH; ++j) {
    float v = row[t * CH + j];
    if (v > mv) { mv = v; mi = t * CH + j; }
  }

  for (int it = 0; it < TOPK; ++it) {
    cv[t] = mv; ci[t] = mi;
    __syncthreads();
    for (int s = 128; s > 0; s >>= 1) {
      if (t < s) {
        float v2 = cv[t + s]; int i2 = ci[t + s];
        if (v2 > cv[t] || (v2 == cv[t] && i2 < ci[t])) { cv[t] = v2; ci[t] = i2; }
      }
      __syncthreads();
    }
    if (t == 0) {
      float v = cv[0];
      topv[b * TOPK + it] = v > 0.f ? v : 0.f;  // ReLU post-activation
      topi[b * TOPK + it] = ci[0];
      sWin = ci[0];
    }
    __syncthreads();
    int widx = sWin;
    if ((widx >> 6) == t) {  // owner chunk rescans
      row[widx] = -__builtin_inff();
      mv = -__builtin_inff();
      mi = t * CH;
      for (int j = 0; j < CH; ++j) {
        float v = row[t * CH + j];
        if (v > mv) { mv = v; mi = t * CH + j; }
      }
    }
    __syncthreads();
  }
}

// ---------------------------------------------------------------------------
// Sparse decode + per-source squared-error partials. One block per row.
// recon[b][sd] = b_dec[sd] + sum_k topv[b][k] * W_dec[topi[b][k]][sd]
// ---------------------------------------------------------------------------
__global__ __launch_bounds__(256) void decode_kernel(const float* __restrict__ x,
                                                     const float* __restrict__ W_dec,
                                                     const float* __restrict__ b_dec,
                                                     const float* __restrict__ topv,
                                                     const int* __restrict__ topi,
                                                     float* __restrict__ recon,
                                                     float* __restrict__ partials) {
  __shared__ float sv[TOPK];
  __shared__ int   si[TOPK];
  __shared__ float r0[256], r1[256], r2[256];
  const int b = blockIdx.x;
  const int t = threadIdx.x;
  if (t < TOPK) { sv[t] = topv[b * TOPK + t]; si[t] = topi[b * TOPK + t]; }
  __syncthreads();

  float s0 = 0.f, s1 = 0.f, s2 = 0.f;
  for (int sd = t; sd < SD; sd += 256) {
    float acc = b_dec[sd];
#pragma unroll 8
    for (int k = 0; k < TOPK; ++k) {
      acc += sv[k] * W_dec[(size_t)si[k] * SD + sd];
    }
    recon[(size_t)b * SD + sd] = acc;
    float d  = x[(size_t)b * SD + sd] - acc;
    float sq = d * d;
    int s = sd >> 10;  // sd / DIN
    if (s == 0) s0 += sq; else if (s == 1) s1 += sq; else s2 += sq;
  }
  r0[t] = s0; r1[t] = s1; r2[t] = s2;
  __syncthreads();
  for (int s = 128; s > 0; s >>= 1) {
    if (t < s) { r0[t] += r0[t + s]; r1[t] += r1[t + s]; r2[t] += r2[t + s]; }
    __syncthreads();
  }
  if (t == 0) {
    partials[b * 3 + 0] = r0[0];
    partials[b * 3 + 1] = r1[0];
    partials[b * 3 + 2] = r2[0];
  }
}

// ---------------------------------------------------------------------------
// Final deterministic reduction of per-row partials -> losses.
// out[0] = main_loss, out[1..3] = source_losses
// ---------------------------------------------------------------------------
__global__ __launch_bounds__(256) void finalize_kernel(const float* __restrict__ partials,
                                                       float* __restrict__ out) {
  __shared__ float r0[256], r1[256], r2[256];
  const int t = threadIdx.x;
  float a0 = 0.f, a1 = 0.f, a2 = 0.f;
  for (int b = t; b < BATCH; b += 256) {
    a0 += partials[b * 3 + 0];
    a1 += partials[b * 3 + 1];
    a2 += partials[b * 3 + 2];
  }
  r0[t] = a0; r1[t] = a1; r2[t] = a2;
  __syncthreads();
  for (int s = 128; s > 0; s >>= 1) {
    if (t < s) { r0[t] += r0[t + s]; r1[t] += r1[t + s]; r2[t] += r2[t + s]; }
    __syncthreads();
  }
  if (t == 0) {
    float src0 = r0[0], src1 = r1[0], src2 = r2[0];
    const float perSrc = 1.0f / ((float)BATCH * (float)DIN);
    out[0] = (src0 + src1 + src2) * (1.0f / ((float)BATCH * (float)SRC * (float)DIN));
    out[1] = src0 * perSrc;
    out[2] = src1 * perSrc;
    out[3] = src2 * perSrc;
  }
}

// ---------------------------------------------------------------------------
extern "C" void kernel_launch(void* const* d_in, const int* in_sizes, int n_in,
                              void* d_out, int out_size, void* d_ws, size_t ws_size,
                              hipStream_t stream) {
  (void)in_sizes; (void)n_in; (void)out_size; (void)ws_size;
  const float* x     = (const float*)d_in[0];   // [B][S][D]
  const float* W_enc = (const float*)d_in[1];   // [S][D][H] == [SD][H]
  const float* W_dec = (const float*)d_in[2];   // [H][S][D] == [H][SD]
  const float* b_enc = (const float*)d_in[3];   // [H]
  const float* b_dec = (const float*)d_in[4];   // [S][D]
  float* out = (float*)d_out;

  // workspace layout
  char* w = (char*)d_ws;
  float* latents = (float*)w;            w += (size_t)BATCH * HID * sizeof(float);   // 256 MB
  __bf16* xb     = (__bf16*)w;           w += (size_t)BATCH * SD * sizeof(__bf16);   // 24 MB
  __bf16* wb     = (__bf16*)w;           w += (size_t)SD * HID * sizeof(__bf16);     // 96 MB
  float* topv    = (float*)w;            w += (size_t)BATCH * TOPK * sizeof(float);
  int*   topi    = (int*)w;              w += (size_t)BATCH * TOPK * sizeof(int);
  float* partials = (float*)w;           w += (size_t)BATCH * 3 * sizeof(float);

  // 1) precision conversion for the compute-bound GEMM (bf16 WMMA path)
  cvt_f32_bf16<<<2048, 256, 0, stream>>>(x, xb, (size_t)BATCH * SD / 4);
  cvt_f32_bf16<<<4096, 256, 0, stream>>>(W_enc, wb, (size_t)SD * HID / 4);

  // 2) encode GEMM (412 GFLOP) on WMMA
  dim3 ggrid(HID / BN, BATCH / BM);  // 128 x 32
  encode_gemm<<<ggrid, 256, 0, stream>>>(xb, wb, b_enc, latents);

  // 3) top-64 + ReLU per row
  topk_kernel<<<BATCH, 256, 0, stream>>>(latents, topv, topi);

  // 4) sparse decode + loss partials (recon goes straight into d_out + 4)
  decode_kernel<<<BATCH, 256, 0, stream>>>(x, W_dec, b_dec, topv, topi,
                                           out + 4, partials);

  // 5) deterministic loss reduction
  finalize_kernel<<<1, 256, 0, stream>>>(partials, out);
}
